// LatentCrossAttention_22943715295777
// MI455X (gfx1250) — compile-verified, matrix operand order checked by reading, not run on hardware
//
#include <hip/hip_runtime.h>
#include <math.h>

// ---------------------------------------------------------------------------
// Latent cross-attention for MI455X (gfx1250, wave32, WMMA bf16 16x16x32).
// Memory-bound (~0.9 GB traffic @ 23.3 TB/s): fuse bias+LN(+softmax) into the
// projection GEMM epilogues, keep intermediates bf16, transposed global
// layouts (KT/VT/KVT) so every producer/consumer moves packed 16-32B chunks,
// and use CDNA5 async global->LDS copies for the conversion-free stagings.
// ---------------------------------------------------------------------------

typedef __bf16 bf16;
typedef __attribute__((ext_vector_type(16))) __bf16 bf16x16;
typedef __attribute__((ext_vector_type(8)))  __bf16 bf16x8;
typedef __attribute__((ext_vector_type(2)))  __bf16 bf16x2;
typedef __attribute__((ext_vector_type(8)))  float  f32x8;

#define EPI_LN    0
#define EPI_LN_SM 1
#define EPI_BIAS  2

__device__ __forceinline__ f32x8 wmma_bf16(bf16x16 a, bf16x16 b, f32x8 c) {
  // D = A(16x32 bf16) * B(32x16 bf16) + C(16x16 f32)
  return __builtin_amdgcn_wmma_f32_16x16x32_bf16(
      false, a, false, b, (short)0, c, false, false);
}

// 16x32 bf16 fragment from LDS (leading dim `ld` elems). ISA 16-bit layout:
// lanes 0-15 hold K=0..7,16..23; lanes 16-31 hold K=8..15,24..31.
__device__ __forceinline__ bf16x16 frag_ld(const bf16* base, int ld) {
  const int lane = threadIdx.x & 31;
  const bf16* p = base + (size_t)(lane & 15) * ld + ((lane >> 4) << 3);
  union { bf16x16 v; bf16x8 h[2]; } u;
  u.h[0] = *(const bf16x8*)(p);
  u.h[1] = *(const bf16x8*)(p + 16);
  return u.v;
}

__device__ __forceinline__ f32x8 f32x8_zero() {
  f32x8 z = {0.f, 0.f, 0.f, 0.f, 0.f, 0.f, 0.f, 0.f};
  return z;
}

// CDNA5 async global->LDS 16B copy (tracked by ASYNCcnt). INST_OFFSET is
// added to BOTH the LDS and the global address, so it works for identity
// copies where both sides share intra-row offsets.
__device__ __forceinline__ void async_copy16(unsigned lds_off, const void* gptr) {
  asm volatile("global_load_async_to_lds_b128 %0, %1, off"
               :: "v"(lds_off), "v"((unsigned long long)(uintptr_t)gptr)
               : "memory");
}
__device__ __forceinline__ void async_copy16_off(unsigned lds_off,
                                                 const void* gptr, int imm) {
  // imm must be a compile-time constant via "n" constraint
  asm volatile("global_load_async_to_lds_b128 %0, %1, off offset:%2"
               :: "v"(lds_off), "v"((unsigned long long)(uintptr_t)gptr),
                  "n"(imm)
               : "memory");
}
__device__ __forceinline__ void wait_async0() {
  asm volatile("s_wait_asynccnt 0x0" ::: "memory");
}
__device__ __forceinline__ unsigned lds_addr32(const void* p) {
  // generic pointer to __shared__: low 32 bits = LDS byte offset (ISA 10.2)
  return (unsigned)(uintptr_t)p;
}

// ---------------------------------------------------------------------------
// Fused projection GEMM: OUT = epilogue(X @ W + bias), X:[Mtot,1024],
// W:[1024,NW] f32. Block = 16 rows x full NW; 8 waves; K-step 32.
// OUT_TR=true stores transposed as OUT[b*NW+col][8192] (contiguous per lane).
// ---------------------------------------------------------------------------
template <int NW, int EPI, bool OUT_BF16, bool X_BF16, bool OUT_TR>
__global__ __launch_bounds__(256) void proj_kernel(
    const void* __restrict__ Xv, const float* __restrict__ W,
    const float* __restrict__ bias, const float* __restrict__ gamma,
    const float* __restrict__ beta, void* __restrict__ OUTv) {
  constexpr int KDIM = 1024;
  constexpr int WCOLS = NW / 8;   // columns per wave
  constexpr int NT = WCOLS / 16;  // 16x16 C tiles per wave
  static_assert(EPI != EPI_LN_SM || NW == 512, "softmax head = wave chunk");

  __shared__ __align__(16) bf16 Alds[16 * 32];
  __shared__ __align__(16) bf16 Blds[32 * NW];
  __shared__ float red_s[8][16];
  __shared__ float red_q[8][16];

  const int t = threadIdx.x;
  const int lane = t & 31;
  const int wave = t >> 5;
  const int row0 = blockIdx.x * 16;

  f32x8 acc[NT];
#pragma unroll
  for (int i = 0; i < NT; ++i) acc[i] = f32x8_zero();

  const float* Xf = (const float*)Xv;
  const bf16* Xb = (const bf16*)Xv;

  for (int k0 = 0; k0 < KDIM; k0 += 32) {
    // prefetch next weight tile (global_prefetch_b8)
    if (k0 + 32 < KDIM)
      __builtin_prefetch(W + (size_t)(k0 + 32) * NW + t * (NW / 64), 0, 0);
    // Stage A tile 16x32 -> bf16 LDS, packed b32 stores
    {
      const int idx = t * 2;
      const int r = idx >> 5, kk = idx & 31;
      if (X_BF16) {
        *(bf16x2*)&Alds[r * 32 + kk] =
            *(const bf16x2*)(Xb + (size_t)(row0 + r) * KDIM + k0 + kk);
      } else {
        const float2 x2 = *(const float2*)(Xf + (size_t)(row0 + r) * KDIM + k0 + kk);
        bf16x2 p = {(bf16)x2.x, (bf16)x2.y};
        *(bf16x2*)&Alds[r * 32 + kk] = p;
      }
    }
    // Stage B tile 32xNW transposed to [n][k]; each thread: 2 k-rows x 4 n
    // -> two float4 global loads + four packed b32 LDS stores.
#pragma unroll
    for (int u0 = 0; u0 < 4 * NW; u0 += 256) {
      const int u = u0 + t;
      const int kk = (u / (NW / 4)) * 2;
      const int n = (u % (NW / 4)) * 4;
      const float4 wa = *(const float4*)(W + (size_t)(k0 + kk) * NW + n);
      const float4 wb = *(const float4*)(W + (size_t)(k0 + kk + 1) * NW + n);
      bf16x2 p0 = {(bf16)wa.x, (bf16)wb.x};
      bf16x2 p1 = {(bf16)wa.y, (bf16)wb.y};
      bf16x2 p2 = {(bf16)wa.z, (bf16)wb.z};
      bf16x2 p3 = {(bf16)wa.w, (bf16)wb.w};
      *(bf16x2*)&Blds[(n + 0) * 32 + kk] = p0;
      *(bf16x2*)&Blds[(n + 1) * 32 + kk] = p1;
      *(bf16x2*)&Blds[(n + 2) * 32 + kk] = p2;
      *(bf16x2*)&Blds[(n + 3) * 32 + kk] = p3;
    }
    __syncthreads();
    const bf16x16 af = frag_ld(Alds, 32);
#pragma unroll
    for (int nt = 0; nt < NT; ++nt) {
      const bf16x16 bfr = frag_ld(Blds + (size_t)(wave * WCOLS + nt * 16) * 32, 32);
      acc[nt] = wmma_bf16(af, bfr, acc[nt]);
    }
    __syncthreads();
  }

  // ---- epilogue (C layout: vgpr i, lane L -> row = i + 8*(L>>4), col = L&15)
  const int half = lane >> 4;
  const int colbase = wave * WCOLS + (lane & 15);

#pragma unroll
  for (int nt = 0; nt < NT; ++nt) {
    const float bcol = bias[colbase + nt * 16];
#pragma unroll
    for (int i = 0; i < 8; ++i) acc[nt][i] += bcol;
  }

  if constexpr (EPI != EPI_BIAS) {
    float s[8], q[8];
#pragma unroll
    for (int i = 0; i < 8; ++i) { s[i] = 0.f; q[i] = 0.f; }
#pragma unroll
    for (int nt = 0; nt < NT; ++nt)
#pragma unroll
      for (int i = 0; i < 8; ++i) {
        const float x = acc[nt][i];
        s[i] += x;
        q[i] += x * x;
      }
#pragma unroll
    for (int m = 1; m < 16; m <<= 1)
#pragma unroll
      for (int i = 0; i < 8; ++i) {
        s[i] += __shfl_xor(s[i], m, 32);
        q[i] += __shfl_xor(q[i], m, 32);
      }
    if ((lane & 15) == 0) {
#pragma unroll
      for (int i = 0; i < 8; ++i) {
        red_s[wave][half * 8 + i] = s[i];
        red_q[wave][half * 8 + i] = q[i];
      }
    }
    __syncthreads();
    float mu[8], rstd[8];
#pragma unroll
    for (int i = 0; i < 8; ++i) {
      const int r = half * 8 + i;
      float ts = 0.f, tq = 0.f;
#pragma unroll
      for (int w = 0; w < 8; ++w) { ts += red_s[w][r]; tq += red_q[w][r]; }
      mu[i] = ts * (1.0f / NW);
      const float var = tq * (1.0f / NW) - mu[i] * mu[i];
      rstd[i] = rsqrtf(var + 1e-5f);
    }
#pragma unroll
    for (int nt = 0; nt < NT; ++nt) {
      const int col = colbase + nt * 16;
      const float g = gamma[col], be = beta[col];
#pragma unroll
      for (int i = 0; i < 8; ++i)
        acc[nt][i] = (acc[nt][i] - mu[i]) * rstd[i] * g + be;
    }
  }

  if constexpr (EPI == EPI_LN_SM) {
    float mx[8];
#pragma unroll
    for (int i = 0; i < 8; ++i) {
      mx[i] = acc[0][i];
#pragma unroll
      for (int nt = 1; nt < NT; ++nt) mx[i] = fmaxf(mx[i], acc[nt][i]);
    }
#pragma unroll
    for (int m = 1; m < 16; m <<= 1)
#pragma unroll
      for (int i = 0; i < 8; ++i) mx[i] = fmaxf(mx[i], __shfl_xor(mx[i], m, 32));
    float sm[8];
#pragma unroll
    for (int i = 0; i < 8; ++i) sm[i] = 0.f;
#pragma unroll
    for (int nt = 0; nt < NT; ++nt)
#pragma unroll
      for (int i = 0; i < 8; ++i) {
        acc[nt][i] = __expf(acc[nt][i] - mx[i]);
        sm[i] += acc[nt][i];
      }
#pragma unroll
    for (int m = 1; m < 16; m <<= 1)
#pragma unroll
      for (int i = 0; i < 8; ++i) sm[i] += __shfl_xor(sm[i], m, 32);
    float rs[8];
#pragma unroll
    for (int i = 0; i < 8; ++i) rs[i] = __builtin_amdgcn_rcpf(sm[i]);
#pragma unroll
    for (int nt = 0; nt < NT; ++nt)
#pragma unroll
      for (int i = 0; i < 8; ++i) acc[nt][i] *= rs[i];
  }

  // ---- store
  if constexpr (OUT_TR) {
    // OUT[(bb*NW + col)][8192]; each lane owns 8 consecutive n -> packed.
    const int bb = row0 >> 13;
    const int n0w = (row0 & 8191) + half * 8;
#pragma unroll
    for (int nt = 0; nt < NT; ++nt) {
      const int col = colbase + nt * 16;
      const size_t base = (size_t)(bb * NW + col) * 8192 + n0w;
      if (OUT_BF16) {
        union { bf16x8 v; bf16 e[8]; } u;
#pragma unroll
        for (int i = 0; i < 8; ++i) u.e[i] = (bf16)acc[nt][i];
        *(bf16x8*)((bf16*)OUTv + base) = u.v;
      } else {
        float* p = (float*)OUTv + base;
        float4 a = {acc[nt][0], acc[nt][1], acc[nt][2], acc[nt][3]};
        float4 b = {acc[nt][4], acc[nt][5], acc[nt][6], acc[nt][7]};
        *(float4*)(p) = a;
        *(float4*)(p + 4) = b;
      }
    }
  } else {
#pragma unroll
    for (int nt = 0; nt < NT; ++nt) {
      const int col = colbase + nt * 16;
#pragma unroll
      for (int i = 0; i < 8; ++i) {
        const size_t off = (size_t)(row0 + half * 8 + i) * NW + col;
        if (OUT_BF16)
          ((bf16*)OUTv)[off] = (bf16)acc[nt][i];
        else
          ((float*)OUTv)[off] = acc[nt][i];
      }
    }
  }
}

// ---------------------------------------------------------------------------
// Column max / sum-exp of K over N=8192, on KT[b*512+c][8192]: one wave per
// column, contiguous float4 streaming, online logsumexp, shfl combine.
// ---------------------------------------------------------------------------
__global__ __launch_bounds__(256) void colstats_kernel(
    const float* __restrict__ KT, float* __restrict__ colmax,
    float* __restrict__ colsum) {
  const int lane = threadIdx.x & 31;
  const int wave = threadIdx.x >> 5;
  const int col = blockIdx.x * 8 + wave;  // [0, 4096)
  const float* base = KT + (size_t)col * 8192;
  float m = -3.0e38f, s = 0.f;
  for (int n = lane * 4; n < 8192; n += 128) {
    const float4 x = *(const float4*)(base + n);
#pragma unroll
    for (int j = 0; j < 4; ++j) {
      const float xv = (j == 0) ? x.x : (j == 1) ? x.y : (j == 2) ? x.z : x.w;
      const float nm = fmaxf(m, xv);
      s = s * __expf(m - nm) + __expf(xv - nm);
      m = nm;
    }
  }
#pragma unroll
  for (int off = 1; off < 32; off <<= 1) {
    const float om = __shfl_xor(m, off, 32);
    const float os = __shfl_xor(s, off, 32);
    const float nm = fmaxf(m, om);
    s = s * __expf(m - nm) + os * __expf(om - nm);
    m = nm;
  }
  if (lane == 0) {
    colmax[col] = m;
    colsum[col] = s;
  }
}

// ---------------------------------------------------------------------------
// kv[b,h] = softmax_cols(K)^T @ V : [64 x 8192] x [8192 x 128].
// KT/VT layouts -> both stagings contiguous; V staging is an identity copy,
// done with async global->LDS b128 (ASYNCcnt). Output KVT[bh][128 d][64 l].
// ---------------------------------------------------------------------------
__global__ __launch_bounds__(256) void kv_kernel(
    const float* __restrict__ KT, const bf16* __restrict__ VT,
    const float* __restrict__ colmax, const float* __restrict__ colsum,
    bf16* __restrict__ KVT) {
  const int bh = blockIdx.x;
  const int b = bh >> 3, h = bh & 7;
  const int t = threadIdx.x, lane = t & 31, wave = t >> 5;
  const int wm = wave >> 2, wn = wave & 3;  // 2 (l) x 4 (d)

  __shared__ __align__(16) bf16 Alds[64 * 32];   // [l][j]
  __shared__ __align__(16) bf16 Blds[128 * 32];  // [d][j]

  const float* Kb = KT + (size_t)(b * 512 + h * 64) * 8192;
  const bf16* Vb = VT + (size_t)(b * 1024 + h * 128) * 8192;

  // per-thread staging coordinates (fixed across the n-loop)
  const int al = t >> 2, aj0 = (t & 3) * 8;    // A: 4 threads per l-row
  const int bd = t >> 1, bj0 = (t & 1) * 16;   // B: 2 threads per d-row
  const float cm = colmax[b * 512 + h * 64 + al];
  const float* Krow = Kb + (size_t)al * 8192 + aj0;
  const bf16* Vrow = Vb + (size_t)bd * 8192 + bj0;
  const unsigned blds_off = lds_addr32(&Blds[bd * 32 + bj0]);

  f32x8 acc[2][2];
#pragma unroll
  for (int mi = 0; mi < 2; ++mi)
#pragma unroll
    for (int ni = 0; ni < 2; ++ni) acc[mi][ni] = f32x8_zero();

  for (int n0 = 0; n0 < 8192; n0 += 32) {
    // async identity copy: 2x16B per thread; same +16B offset on both sides
    async_copy16(blds_off, Vrow + n0);
    async_copy16_off(blds_off, Vrow + n0, 16);
    // exp-weighted K staging (needs conversion -> regular path)
    const float4 xa = *(const float4*)(Krow + n0);
    const float4 xb = *(const float4*)(Krow + n0 + 4);
    union { bf16x8 v; bf16 e[8]; } ua;
    ua.e[0] = (bf16)__expf(xa.x - cm);
    ua.e[1] = (bf16)__expf(xa.y - cm);
    ua.e[2] = (bf16)__expf(xa.z - cm);
    ua.e[3] = (bf16)__expf(xa.w - cm);
    ua.e[4] = (bf16)__expf(xb.x - cm);
    ua.e[5] = (bf16)__expf(xb.y - cm);
    ua.e[6] = (bf16)__expf(xb.z - cm);
    ua.e[7] = (bf16)__expf(xb.w - cm);
    *(bf16x8*)&Alds[al * 32 + aj0] = ua.v;
    wait_async0();
    __syncthreads();
#pragma unroll
    for (int mi = 0; mi < 2; ++mi) {
      const bf16x16 a = frag_ld(Alds + (size_t)(wm * 32 + mi * 16) * 32, 32);
#pragma unroll
      for (int ni = 0; ni < 2; ++ni) {
        const bf16x16 bb = frag_ld(Blds + (size_t)(wn * 32 + ni * 16) * 32, 32);
        acc[mi][ni] = wmma_bf16(a, bb, acc[mi][ni]);
      }
    }
    __syncthreads();
  }

  const int half = lane >> 4;
#pragma unroll
  for (int mi = 0; mi < 2; ++mi) {
    const int lb = wm * 32 + mi * 16 + half * 8;  // 8 consecutive l per lane
    float r[8];
#pragma unroll
    for (int i = 0; i < 8; ++i)
      r[i] = __builtin_amdgcn_rcpf(colsum[b * 512 + h * 64 + lb + i]);
#pragma unroll
    for (int ni = 0; ni < 2; ++ni) {
      const int d = wn * 32 + ni * 16 + (lane & 15);
      union { bf16x8 v; bf16 e[8]; } u;
#pragma unroll
      for (int i = 0; i < 8; ++i) u.e[i] = (bf16)(acc[mi][ni][i] * r[i]);
      *(bf16x8*)&KVT[((size_t)bh * 128 + d) * 64 + lb] = u.v;
    }
  }
}

// ---------------------------------------------------------------------------
// attn (transposed form): out^T[d,m] = kv^T[d,l] @ Q^T[l,m] per (b,h).
// A = KVT (flat async copy), B = Q rows (async row copies); stores to the
// standard ATT[m][c] layout are contiguous bf16x8 per lane.
// ---------------------------------------------------------------------------
__global__ __launch_bounds__(256) void attn_kernel(
    const bf16* __restrict__ Q, const bf16* __restrict__ KVT,
    bf16* __restrict__ ATT) {
  const int bh = blockIdx.x;
  const int b = bh >> 3, h = bh & 7;
  const int t = threadIdx.x, lane = t & 31, wave = t >> 5;
  const int wm = wave >> 1, wn = wave & 1;  // 4 (d) x 2 (m)

  __shared__ __align__(16) bf16 Alds[128 * 64];  // [d][l]
  __shared__ __align__(16) bf16 Blds[256 * 64];  // [m][l]

  // A: flat 16KB identity copy KVT[bh] -> Alds (4x16B per thread, stride 4KB)
  const bf16* KVTb = KVT + (size_t)bh * 128 * 64;
  {
    const unsigned la = lds_addr32(&Alds[t * 8]);
    const bf16* ga = KVTb + t * 8;
    async_copy16(la, ga);
    async_copy16_off(la, ga, 4096);
    async_copy16_off(la, ga, 8192);
    async_copy16_off(la, ga, 12288);
  }
  // B: one 128B row per thread, Q natural layout -> Blds[m][l]
  const bf16* Qb = Q + (size_t)b * 256 * 512 + h * 64;
  {
    const unsigned lb = lds_addr32(&Blds[t * 64]);
    const bf16* gb = Qb + (size_t)t * 512;
    async_copy16(lb, gb);
    async_copy16_off(lb, gb, 16);
    async_copy16_off(lb, gb, 32);
    async_copy16_off(lb, gb, 48);
    async_copy16_off(lb, gb, 64);
    async_copy16_off(lb, gb, 80);
    async_copy16_off(lb, gb, 96);
    async_copy16_off(lb, gb, 112);
  }
  wait_async0();
  __syncthreads();

  f32x8 acc[2][8];
#pragma unroll
  for (int mi = 0; mi < 2; ++mi)
#pragma unroll
    for (int ni = 0; ni < 8; ++ni) acc[mi][ni] = f32x8_zero();

#pragma unroll
  for (int ks = 0; ks < 2; ++ks) {
    bf16x16 a[2];
#pragma unroll
    for (int mi = 0; mi < 2; ++mi)
      a[mi] = frag_ld(Alds + (size_t)(wm * 32 + mi * 16) * 64 + ks * 32, 64);
#pragma unroll
    for (int ni = 0; ni < 8; ++ni) {
      const bf16x16 bb =
          frag_ld(Blds + (size_t)(wn * 128 + ni * 16) * 64 + ks * 32, 64);
#pragma unroll
      for (int mi = 0; mi < 2; ++mi) acc[mi][ni] = wmma_bf16(a[mi], bb, acc[mi][ni]);
    }
  }

  const int half = lane >> 4;
  bf16* Ab = ATT + (size_t)b * 256 * 1024 + h * 128;
#pragma unroll
  for (int mi = 0; mi < 2; ++mi) {
    const int db = wm * 32 + mi * 16 + half * 8;  // 8 consecutive d per lane
#pragma unroll
    for (int ni = 0; ni < 8; ++ni) {
      const int m = wn * 128 + ni * 16 + (lane & 15);
      union { bf16x8 v; bf16 e[8]; } u;
#pragma unroll
      for (int i = 0; i < 8; ++i) u.e[i] = (bf16)acc[mi][ni][i];
      *(bf16x8*)(Ab + (size_t)m * 1024 + db) = u.v;
    }
  }
}

// ---------------------------------------------------------------------------
// Host-side orchestration. Inputs (setup_inputs order):
//  0 input_tokens 1 query_tokens 2 q_w 3 q_b 4 q_g 5 q_beta
//  6 k_w 7 k_b 8 k_g 9 k_beta 10 v_w 11 v_b 12 v_g 13 v_beta 14 proj_w 15 proj_b
// ---------------------------------------------------------------------------
extern "C" void kernel_launch(void* const* d_in, const int* in_sizes, int n_in,
                              void* d_out, int out_size, void* d_ws,
                              size_t ws_size, hipStream_t stream) {
  (void)in_sizes; (void)n_in; (void)out_size; (void)ws_size;
  const float* x   = (const float*)d_in[0];
  const float* xq  = (const float*)d_in[1];
  const float* q_w = (const float*)d_in[2];
  const float* q_b = (const float*)d_in[3];
  const float* q_g = (const float*)d_in[4];
  const float* q_be= (const float*)d_in[5];
  const float* k_w = (const float*)d_in[6];
  const float* k_b = (const float*)d_in[7];
  const float* k_g = (const float*)d_in[8];
  const float* k_be= (const float*)d_in[9];
  const float* v_w = (const float*)d_in[10];
  const float* v_b = (const float*)d_in[11];
  const float* v_g = (const float*)d_in[12];
  const float* v_be= (const float*)d_in[13];
  const float* p_w = (const float*)d_in[14];
  const float* p_b = (const float*)d_in[15];

  char* ws = (char*)d_ws;
  const size_t SZ_K  = (size_t)8 * 8192 * 512 * sizeof(float);  // KT 128 MB
  const size_t SZ_V  = (size_t)8 * 8192 * 1024 * sizeof(bf16);  // VT 128 MB
  const size_t SZ_CM = (size_t)8 * 512 * sizeof(float);
  const size_t SZ_QS = (size_t)8 * 256 * 512 * sizeof(bf16);
  const size_t SZ_KV = (size_t)64 * 64 * 128 * sizeof(bf16);
  size_t off = 0;
  float* KT   = (float*)(ws + off); off += SZ_K;
  bf16*  VT   = (bf16*)(ws + off);  off += SZ_V;
  float* cmax = (float*)(ws + off); off += SZ_CM;
  float* csum = (float*)(ws + off); off += SZ_CM;
  bf16*  Qs   = (bf16*)(ws + off);  off += SZ_QS;
  bf16*  KVT  = (bf16*)(ws + off);  off += SZ_KV;
  bf16*  ATT  = (bf16*)(ws + off);

  // 1) KT = LN(X @ k_w + k_b)^T            [b*512+c][8192] f32
  proj_kernel<512, EPI_LN, false, false, true><<<65536 / 16, 256, 0, stream>>>(
      x, k_w, k_b, k_g, k_be, KT);
  // 2) VT = LN(X @ v_w + v_b)^T            [b*1024+c][8192] bf16
  proj_kernel<1024, EPI_LN, true, false, true><<<65536 / 16, 256, 0, stream>>>(
      x, v_w, v_b, v_g, v_be, VT);
  // 3) Q = softmax64(LN(Xq @ q_w + q_b))   [2048 x 512] bf16 (natural)
  proj_kernel<512, EPI_LN_SM, true, false, false><<<2048 / 16, 256, 0, stream>>>(
      xq, q_w, q_b, q_g, q_be, Qs);
  // 4) per-column max / sum-exp of K over N
  colstats_kernel<<<4096 / 8, 256, 0, stream>>>(KT, cmax, csum);
  // 5) kv^T per head                        KVT[bh][128][64] bf16
  kv_kernel<<<64, 256, 0, stream>>>(KT, VT, cmax, csum, KVT);
  // 6) attn = Q @ kv per head               ATT[2048 x 1024] bf16
  attn_kernel<<<64, 256, 0, stream>>>(Qs, KVT, ATT);
  // 7) out = attn @ proj_w + proj_b         [2048 x 1024] f32
  proj_kernel<1024, EPI_BIAS, false, true, false><<<2048 / 16, 256, 0, stream>>>(
      ATT, p_w, p_b, nullptr, nullptr, (float*)d_out);
}
